// TopM_MHSA_48679159333601
// MI455X (gfx1250) — compile-verified, matrix operand order checked by reading, not run on hardware
//
#include <hip/hip_runtime.h>
#include <hip/hip_bf16.h>

// ---------------------------------------------------------------------------
// MI455X (gfx1250) 4-layer ViT encoder forward.
// f16 WMMA (V_WMMA_F32_16X16X32_F16), f32 accumulate. Weights pre-transposed
// (Wt[N][K] f16) so BOTH GEMM tiles are plain row-major tiles staged by the
// Tensor Data Mover into double-buffered LDS (tensor_load_to_lds +
// s_wait_tensorcnt), pipelined against the WMMAs. Attention exploits the
// post-softmax ">0.5" mask: at most one prob per row survives, so
// out = (Z<2) ? v[argmax]/Z : 0 -> single pass, no P*V matmul.
// ---------------------------------------------------------------------------

#define Bb 4
#define Nn 1024
#define Cc 1024
#define Hh 16
#define HDd 64
#define FFf 4096
#define Mtot (Bb * Nn)
#define EPS_ 1e-5f

typedef _Float16 v16h __attribute__((ext_vector_type(16)));
typedef float    v8f  __attribute__((ext_vector_type(8)));
typedef unsigned int uintx4 __attribute__((ext_vector_type(4)));
typedef int intx4 __attribute__((ext_vector_type(4)));
typedef int intx8 __attribute__((ext_vector_type(8)));

union F16x16 { uint4 q[2]; v16h v; };
union F16x32 { uint4 q[4]; _Float16 h[32]; };

static __device__ __forceinline__ v8f wmma_f16(v16h a, v16h b, v8f c) {
  return __builtin_amdgcn_wmma_f32_16x16x32_f16(false, a, false, b, (short)0, c,
                                                false, false);
}

// A-style fragment load: lane holds row (lane&15); K pairs {0..7,16..23} for
// lanes 0-15, {8..15,24..31} for lanes 16-31 => two 16B loads per lane.
static __device__ __forceinline__ v16h load_frag(const _Float16* rowbase, int lane) {
  const int o0 = (lane & 16) ? 8 : 0;   // in halfs
  F16x16 u;
  u.q[0] = *(const uint4*)(rowbase + o0);
  u.q[1] = *(const uint4*)(rowbase + o0 + 16);
  return u.v;
}

// ---------------------------------------------------------------------------
// TDM: async tensor load of a rows x 32 f16 tile (row stride ld elements)
// into LDS with hardware padding (16 DWORDs data + 4 DWORDs pad = 40-half
// stride). D# per CDNA5 ISA ch.8.
// ---------------------------------------------------------------------------
#if __has_builtin(__builtin_amdgcn_tensor_load_to_lds)
#define USE_TDM 1
#endif

#ifdef USE_TDM
static __device__ __forceinline__ void tdm_load_tile(const _Float16* g,
                                                     unsigned int ldsoff,
                                                     int rows, int ld) {
  const unsigned long long ga = (unsigned long long)(uintptr_t)g;
  uintx4 g0;
  g0.x = 1u;                                         // count=1 (valid D#)
  g0.y = ldsoff;                                     // lds_addr (bytes)
  g0.z = (unsigned int)(ga & 0xffffffffu);           // global_addr[31:0]
  g0.w = (unsigned int)((ga >> 32) & 0x1ffffffu)     // global_addr[56:32]
         | (2u << 30);                               // type=2 ("image")
  intx8 g1;
  g1[0] = (int)((1u << 16)          // data_size=1 -> 2 bytes
                | (1u << 20)        // pad_enable
                | (3u << 22)        // pad_interval: 16 DWORDs (64B row)
                | (3u << 25));      // pad_amount: 4 DWORDs (8 halfs)
  g1[1] = (int)(32u << 16);         // tensor_dim0 = 32
  g1[2] = (int)((unsigned)rows << 16); // tensor_dim1
  g1[3] = (int)(32u << 16);         // tile_dim0 = 32
  g1[4] = rows;                     // tile_dim1
  g1[5] = ld;                       // tensor_dim0_stride (elements)
  g1[6] = 0;
  g1[7] = 0;
  const intx4 z4 = {0, 0, 0, 0};
#if defined(__clang_major__) && (__clang_major__ >= 23)
  const intx8 z8 = {0, 0, 0, 0, 0, 0, 0, 0};
  __builtin_amdgcn_tensor_load_to_lds(g0, g1, z4, z4, z8, 0);
#else
  __builtin_amdgcn_tensor_load_to_lds(g0, g1, z4, z4, 0);
#endif
}
#endif

// ---------------------------------------------------------------------------
// Elementwise kernels
// ---------------------------------------------------------------------------
__global__ __launch_bounds__(256) void k_add_pos(float* __restrict__ out,
                                                 const float* __restrict__ x,
                                                 const float* __restrict__ pos,
                                                 int total, int mask) {
  int i = blockIdx.x * 256 + threadIdx.x;
  if (i < total) out[i] = x[i] + pos[i & mask];
}

// f32 [K][Nc] -> f16 transposed [Nc][K], 32x32 LDS tile.
__global__ __launch_bounds__(256) void k_cvt_tr(_Float16* __restrict__ dstT,
                                                const float* __restrict__ src,
                                                int K, int Nc) {
  __shared__ _Float16 Tt[32][36];
  const int t = threadIdx.x;
  const int n0 = blockIdx.x * 32, k0 = blockIdx.y * 32;
  {
    const int r = t >> 3, c4 = (t & 7) * 4;      // k-row, n-col group
    const float4 f = *(const float4*)(src + (size_t)(k0 + r) * Nc + n0 + c4);
    Tt[c4 + 0][r] = (_Float16)f.x;
    Tt[c4 + 1][r] = (_Float16)f.y;
    Tt[c4 + 2][r] = (_Float16)f.z;
    Tt[c4 + 3][r] = (_Float16)f.w;
  }
  __syncthreads();
  {
    const int n = t >> 3, kc = (t & 7) * 4;
    union { unsigned long long u; _Float16 h[4]; } o;
    o.h[0] = Tt[n][kc + 0]; o.h[1] = Tt[n][kc + 1];
    o.h[2] = Tt[n][kc + 2]; o.h[3] = Tt[n][kc + 3];
    *(unsigned long long*)(dstT + (size_t)(n0 + n) * K + k0 + kc) = o.u;
  }
}

__global__ __launch_bounds__(256) void k_ln(_Float16* __restrict__ dst,
                                            const float* __restrict__ src,
                                            const float* __restrict__ g,
                                            const float* __restrict__ bta) {
  __shared__ float red[256];
  const int row = blockIdx.x, t = threadIdx.x;
  const float* p = src + (size_t)row * Cc;
  float v0 = p[t], v1 = p[t + 256], v2 = p[t + 512], v3 = p[t + 768];
  red[t] = v0 + v1 + v2 + v3;
  __syncthreads();
  for (int o = 128; o > 0; o >>= 1) { if (t < o) red[t] += red[t + o]; __syncthreads(); }
  const float mean = red[0] * (1.0f / Cc);
  __syncthreads();
  float d0 = v0 - mean, d1 = v1 - mean, d2 = v2 - mean, d3 = v3 - mean;
  red[t] = d0 * d0 + d1 * d1 + d2 * d2 + d3 * d3;
  __syncthreads();
  for (int o = 128; o > 0; o >>= 1) { if (t < o) red[t] += red[t + o]; __syncthreads(); }
  const float rstd = rsqrtf(red[0] * (1.0f / Cc) + EPS_);
  _Float16* q = dst + (size_t)row * Cc;
  q[t]       = (_Float16)(d0 * rstd * g[t]       + bta[t]);
  q[t + 256] = (_Float16)(d1 * rstd * g[t + 256] + bta[t + 256]);
  q[t + 512] = (_Float16)(d2 * rstd * g[t + 512] + bta[t + 512]);
  q[t + 768] = (_Float16)(d3 * rstd * g[t + 768] + bta[t + 768]);
}

// ---------------------------------------------------------------------------
// Tiled WMMA GEMM: C[M,Nc] = A16[M,K] x Wt16[Nc,K]^T + bias.
// 128 thr (4 waves), tile 128x64, BK=32, double-buffered TDM pipeline.
// MODE: 0=f16 store, 1=f16+exact GELU, 2=f32 residual accumulate.
// ---------------------------------------------------------------------------
#define ASTR 40   // padded LDS row stride (halfs)
template <int MODE>
__global__ __launch_bounds__(128) void k_gemm(const _Float16* __restrict__ A,
                                              const _Float16* __restrict__ Wt,
                                              const float* __restrict__ bias,
                                              float* __restrict__ outAcc,
                                              _Float16* __restrict__ out16,
                                              int K, int Nc) {
  __shared__ _Float16 As[2 * 128 * ASTR];
  __shared__ _Float16 Bs[2 * 64 * ASTR];
  const int t = threadIdx.x, lane = t & 31, wv = t >> 5;
  const int n0 = blockIdx.x * 64, m0 = blockIdx.y * 128;
  const _Float16* Ab = A + (size_t)m0 * K;
  const _Float16* Bb_ = Wt + (size_t)n0 * K;
  const int nk = K / 32;

  v8f acc[2][4];
#pragma unroll
  for (int s = 0; s < 2; ++s)
#pragma unroll
    for (int c = 0; c < 4; ++c) acc[s][c] = (v8f){0, 0, 0, 0, 0, 0, 0, 0};

#ifdef USE_TDM
  const unsigned int As_off = (unsigned int)(uintptr_t)(&As[0]);
  const unsigned int Bs_off = (unsigned int)(uintptr_t)(&Bs[0]);
  if (wv == 0) {               // prologue: fill buffer 0
    tdm_load_tile(Ab, As_off, 128, K);
    tdm_load_tile(Bb_, Bs_off, 64, K);
  }
#endif

  for (int i = 0; i < nk; ++i) {
#ifdef USE_TDM
    const int cur = i & 1;
    __syncthreads();           // everyone done with buffer cur^1
    if (wv == 0) {
      if (i + 1 < nk) {        // prefetch next k-step into the other buffer
        const int nx = cur ^ 1;
        tdm_load_tile(Ab + (i + 1) * 32, As_off + nx * (128 * ASTR * 2), 128, K);
        tdm_load_tile(Bb_ + (i + 1) * 32, Bs_off + nx * (64 * ASTR * 2), 64, K);
        __builtin_amdgcn_s_wait_tensorcnt(2);   // current buffer complete
      } else {
        __builtin_amdgcn_s_wait_tensorcnt(0);
      }
    }
    __syncthreads();           // broadcast completion
#else
    const int cur = 0;
    __syncthreads();
    {   // manual A tile: one 32-half row per thread
      const uint4* s = (const uint4*)(Ab + (size_t)t * K + i * 32);
      uint4* d = (uint4*)(As + t * ASTR);
      d[0] = s[0]; d[1] = s[1]; d[2] = s[2]; d[3] = s[3];
    }
    {   // manual B tile: straight rows of Wt
      const int r = t >> 1, h2 = (t & 1) * 16;
      const uint4* s = (const uint4*)(Bb_ + (size_t)r * K + i * 32 + h2);
      uint4* d = (uint4*)(Bs + r * ASTR + h2);
      d[0] = s[0]; d[1] = s[1];
    }
    __syncthreads();
#endif
    const _Float16* Ac = As + cur * (128 * ASTR);
    const _Float16* Bc = Bs + cur * (64 * ASTR);
    const v16h a0 = load_frag(Ac + (wv * 32 + (lane & 15)) * ASTR, lane);
    const v16h a1 = load_frag(Ac + (wv * 32 + 16 + (lane & 15)) * ASTR, lane);
#pragma unroll
    for (int c = 0; c < 4; ++c) {
      const v16h fb = load_frag(Bc + (c * 16 + (lane & 15)) * ASTR, lane);
      acc[0][c] = wmma_f16(a0, fb, acc[0][c]);
      acc[1][c] = wmma_f16(a1, fb, acc[1][c]);
    }
  }

  const int nbase = n0 + (lane & 15);
#pragma unroll
  for (int s = 0; s < 2; ++s) {
    const int mbase = m0 + wv * 32 + s * 16 + ((lane & 16) ? 8 : 0);
#pragma unroll
    for (int c = 0; c < 4; ++c) {
      const int n = nbase + c * 16;
      const float bv = bias[n];
#pragma unroll
      for (int r = 0; r < 8; ++r) {
        float v = acc[s][c][r] + bv;
        if (MODE == 1) v = 0.5f * v * (1.0f + erff(v * 0.70710678118654752f));
        if (MODE == 2) outAcc[(size_t)(mbase + r) * Nc + n] += v;
        else           out16[(size_t)(mbase + r) * Nc + n] = (_Float16)v;
      }
    }
  }
}

// ---------------------------------------------------------------------------
// Top-1 masked attention (single pass, no P*V matmul).
// ---------------------------------------------------------------------------
#define SBSTR 36   // Sb row stride (f32)
__global__ __launch_bounds__(256) void k_attn(const _Float16* __restrict__ qkv,
                                              _Float16* __restrict__ aout) {
  __shared__ float Sb[8][16 * SBSTR];
  __shared__ float Pm[8][32];
  __shared__ float Pz[8][32];
  __shared__ int   Pi[8][32];
  const int t = threadIdx.x, lane = t & 31, wv = t >> 5;
  const int b = blockIdx.x >> 4, h = blockIdx.x & 15;
  const int qt = blockIdx.y * 8 + wv;

  // Q fragments, pre-scaled by SCALE = 0.125 (exact in f16)
  const size_t qoff = ((size_t)(b * Nn + qt * 16 + (lane & 15))) * (3 * Cc) + h * HDd;
  v16h aq0 = load_frag(qkv + qoff, lane) * (_Float16)0.125f;
  v16h aq1 = load_frag(qkv + qoff + 32, lane) * (_Float16)0.125f;

  float rm = -3.0e38f, rz = 0.0f;
  int ri = 0;
  const int row = lane & 15, colb = (lane >> 4) * 16;   // owner-lane role
  float* sbw = Sb[wv];

  for (int c0 = 0; c0 < Nn; c0 += 32) {
    const size_t kr0 = ((size_t)(b * Nn + c0 + (lane & 15))) * (3 * Cc) + Cc + h * HDd;
    const size_t kr1 = kr0 + (size_t)16 * (3 * Cc);
    v8f S0 = (v8f){0, 0, 0, 0, 0, 0, 0, 0}, S1 = S0;
    S0 = wmma_f16(aq0, load_frag(qkv + kr0, lane), S0);
    S0 = wmma_f16(aq1, load_frag(qkv + kr0 + 32, lane), S0);
    S1 = wmma_f16(aq0, load_frag(qkv + kr1, lane), S1);
    S1 = wmma_f16(aq1, load_frag(qkv + kr1 + 32, lane), S1);

    const int mb = (lane & 16) ? 8 : 0, kc = lane & 15;
#pragma unroll
    for (int r = 0; r < 8; ++r) {
      sbw[(r + mb) * SBSTR + kc]      = S0[r];
      sbw[(r + mb) * SBSTR + 16 + kc] = S1[r];
    }

    // owner-lane reduction, pairwise trees (depth 4, good ILP)
    const float4* pr = (const float4*)(sbw + row * SBSTR + colb);
    const float4 f0 = pr[0], f1 = pr[1], f2 = pr[2], f3 = pr[3];
    float v[16] = {f0.x, f0.y, f0.z, f0.w, f1.x, f1.y, f1.z, f1.w,
                   f2.x, f2.y, f2.z, f2.w, f3.x, f3.y, f3.z, f3.w};
    float mv[8]; int mi[8];
#pragma unroll
    for (int j = 0; j < 8; ++j) {
      const bool g = v[2 * j + 1] > v[2 * j];
      mv[j] = g ? v[2 * j + 1] : v[2 * j];
      mi[j] = g ? 2 * j + 1 : 2 * j;
    }
#pragma unroll
    for (int j = 0; j < 4; ++j) {
      const bool g = mv[2 * j + 1] > mv[2 * j];
      mv[j] = g ? mv[2 * j + 1] : mv[2 * j];
      mi[j] = g ? mi[2 * j + 1] : mi[2 * j];
    }
    { const bool g = mv[1] > mv[0]; mv[0] = g ? mv[1] : mv[0]; mi[0] = g ? mi[1] : mi[0]; }
    { const bool g = mv[3] > mv[2]; mv[2] = g ? mv[3] : mv[2]; mi[2] = g ? mi[3] : mi[2]; }
    const bool gg = mv[2] > mv[0];
    const float tm = gg ? mv[2] : mv[0];
    const int   ta = gg ? mi[2] : mi[0];

    const float nm = fmaxf(rm, tm);
    float e[16];
#pragma unroll
    for (int j = 0; j < 16; ++j) e[j] = __expf(v[j] - nm);
#pragma unroll
    for (int j = 0; j < 8; ++j) e[j] += e[j + 8];
#pragma unroll
    for (int j = 0; j < 4; ++j) e[j] += e[j + 4];
    const float zs = (e[0] + e[2]) + (e[1] + e[3]);
    rz = rz * __expf(rm - nm) + zs;
    ri = (tm > rm) ? (c0 + colb + ta) : ri;
    rm = nm;
  }

  // merge the two key-half partials per row (online-softmax merge)
  Pm[wv][lane] = rm; Pz[wv][lane] = rz; Pi[wv][lane] = ri;
  const float m0 = Pm[wv][row], m1 = Pm[wv][row + 16];
  const float z0 = Pz[wv][row], z1 = Pz[wv][row + 16];
  const int   i0 = Pi[wv][row], i1 = Pi[wv][row + 16];
  const float mf = fmaxf(m0, m1);
  const float zf = z0 * __expf(m0 - mf) + z1 * __expf(m1 - mf);
  const int   ix = (m0 >= m1) ? i0 : i1;
  const float fs = (zf < 2.0f) ? (1.0f / zf) : 0.0f;   // p_max>0.5 <=> Z<2

  const int dh = (lane >> 4) * 32;
  const _Float16* pv = qkv + ((size_t)(b * Nn + ix)) * (3 * Cc) + 2 * Cc + h * HDd + dh;
  F16x32 uv, ov;
  const uint4* ps = (const uint4*)pv;
  uv.q[0] = ps[0]; uv.q[1] = ps[1]; uv.q[2] = ps[2]; uv.q[3] = ps[3];
#pragma unroll
  for (int j = 0; j < 32; ++j) ov.h[j] = (_Float16)((float)uv.h[j] * fs);
  uint4* po = (uint4*)(aout + ((size_t)(b * Nn + qt * 16 + row)) * Cc + h * HDd + dh);
  po[0] = ov.q[0]; po[1] = ov.q[1]; po[2] = ov.q[2]; po[3] = ov.q[3];
}

// ---------------------------------------------------------------------------
// Host orchestration
// ---------------------------------------------------------------------------
extern "C" void kernel_launch(void* const* d_in, const int* in_sizes, int n_in,
                              void* d_out, int out_size, void* d_ws, size_t ws_size,
                              hipStream_t stream) {
  (void)in_sizes; (void)n_in; (void)out_size; (void)ws_size;
  const float* x     = (const float*)d_in[0];
  const float* pos   = (const float*)d_in[1];
  const float* qkv_w = (const float*)d_in[2];
  const float* qkv_b = (const float*)d_in[3];
  const float* proj_w= (const float*)d_in[4];
  const float* proj_b= (const float*)d_in[5];
  const float* ln1_g = (const float*)d_in[6];
  const float* ln1_b = (const float*)d_in[7];
  const float* ln2_g = (const float*)d_in[8];
  const float* ln2_b = (const float*)d_in[9];
  const float* fc1_w = (const float*)d_in[10];
  const float* fc1_b = (const float*)d_in[11];
  const float* fc2_w = (const float*)d_in[12];
  const float* fc2_b = (const float*)d_in[13];
  float* out = (float*)d_out;
  char* ws = (char*)d_ws;

  _Float16* buf0 = (_Float16*)(ws);                      // 32MB: qkv16 | fc1out16
  _Float16* buf1 = (_Float16*)(ws + (32ull << 20));      //  8MB: xn16 | attnout16
  _Float16* wq   = (_Float16*)(ws + (40ull << 20));      //  6MB  Wt [3C][C]
  _Float16* wp   = (_Float16*)(ws + (46ull << 20));      //  2MB  Wt [C][C]
  _Float16* w1   = (_Float16*)(ws + (48ull << 20));      //  8MB  Wt [FF][C]
  _Float16* w2   = (_Float16*)(ws + (56ull << 20));      //  8MB  Wt [C][FF]
  const int total = Mtot * Cc;
  k_add_pos<<<(total + 255) / 256, 256, 0, stream>>>(out, x, pos, total, Nn * Cc - 1);

  for (int i = 0; i < 4; ++i) {
    // transpose+convert this layer's weights: Wt[N][K] f16
    k_cvt_tr<<<dim3(3 * Cc / 32, Cc / 32), 256, 0, stream>>>(wq, qkv_w + (size_t)i * Cc * 3 * Cc, Cc, 3 * Cc);
    k_cvt_tr<<<dim3(Cc / 32, Cc / 32), 256, 0, stream>>>(wp, proj_w + (size_t)i * Cc * Cc, Cc, Cc);
    k_cvt_tr<<<dim3(FFf / 32, Cc / 32), 256, 0, stream>>>(w1, fc1_w + (size_t)i * Cc * FFf, Cc, FFf);
    k_cvt_tr<<<dim3(Cc / 32, FFf / 32), 256, 0, stream>>>(w2, fc2_w + (size_t)i * FFf * Cc, FFf, Cc);

    k_ln<<<Mtot, 256, 0, stream>>>(buf1, out, ln1_g + i * Cc, ln1_b + i * Cc);
    k_gemm<0><<<dim3(3 * Cc / 64, Mtot / 128), 128, 0, stream>>>(
        buf1, wq, qkv_b + (size_t)i * 3 * Cc, nullptr, buf0, Cc, 3 * Cc);
    k_attn<<<dim3(Bb * Hh, 8), 256, 0, stream>>>(buf0, buf1);
    k_gemm<2><<<dim3(Cc / 64, Mtot / 128), 128, 0, stream>>>(
        buf1, wp, proj_b + (size_t)i * Cc, out, nullptr, Cc, Cc);
    k_ln<<<Mtot, 256, 0, stream>>>(buf1, out, ln2_g + i * Cc, ln2_b + i * Cc);
    k_gemm<1><<<dim3(FFf / 64, Mtot / 128), 128, 0, stream>>>(
        buf1, w1, fc1_b + (size_t)i * FFf, nullptr, buf0, Cc, FFf);
    k_gemm<2><<<dim3(Cc / 64, Mtot / 128), 128, 0, stream>>>(
        buf0, w2, fc2_b + (size_t)i * Cc, out, nullptr, FFf, Cc);
  }
}
